// Filterbank_21655225106961
// MI455X (gfx1250) — matrix operand order repaired and checked by reading, not compiled, not run on hardware
//
#include <hip/hip_runtime.h>
#include <hip/hip_bf16.h>
#include <math.h>

typedef __attribute__((ext_vector_type(16))) __bf16 v16bf;
typedef __attribute__((ext_vector_type(8)))  float  v8f;

#define N_FILT   80
#define FILT_DIM 251
#define KPAD     256           // 251 padded to 8 x 32 for bf16 WMMA K-steps
#define SEQ      32000
#define PAD_L    125
#define NTILE_WG 256           // 8 waves x 2 tiles x 16 columns
#define WIN      512           // 256 cols + 250 halo + K-pad slack
#define FS_F     16000.0f

// ---------------------------------------------------------------------------
// Kernel 1: synthesize the 80x251 sinc filters (Hamming-windowed, max-norm),
// store as bf16, zero-padded to 80x256 so padded-K WMMA taps contribute 0.
// One block per filter, 256 threads (one per padded tap).
// ---------------------------------------------------------------------------
__global__ __launch_bounds__(256) void gen_filters(const float* __restrict__ fb1,
                                                   const float* __restrict__ fband,
                                                   __bf16* __restrict__ filt) {
    const int m = blockIdx.x;
    const int j = threadIdx.x;

    __shared__ float red[256];

    const float min_f = 50.0f / FS_F;
    const float beg = fabsf(fb1[m]) + min_f;
    const float end = beg + fabsf(fband[m]) + min_f;
    const float two_pi = 6.28318530717958647692f;

    float bp = 0.0f;
    if (j < FILT_DIM) {
        if (j == PAD_L) {
            bp = 2.0f * end - 2.0f * beg;         // sinc(0) == 1 for both terms
        } else {
            float t = fabsf((float)(j - PAD_L));  // 1..125, sinc is even
            float ae = two_pi * (end * FS_F) * t;
            float ab = two_pi * (beg * FS_F) * t;
            bp = 2.0f * end * (sinf(ae) / ae) - 2.0f * beg * (sinf(ab) / ab);
        }
    }

    // block max-reduce over the 251 valid taps
    red[j] = (j < FILT_DIM) ? bp : -INFINITY;
    __syncthreads();
    #pragma unroll
    for (int s = 128; s > 0; s >>= 1) {
        if (j < s) red[j] = fmaxf(red[j], red[j + s]);
        __syncthreads();
    }
    const float mx = red[0];

    float val = 0.0f;
    if (j < FILT_DIM) {
        // window = 0.54 - 0.46*cos(2*pi*j/250)  (n = j*251/250, /251)
        float w = 0.54f - 0.46f * cosf(two_pi * (float)j / 250.0f);
        val = (bp / mx) * w;
    }
    filt[(size_t)m * KPAD + j] = (__bf16)val;     // j in [251,256) writes 0
}

// ---------------------------------------------------------------------------
// Kernel 2: implicit-GEMM conv via v_wmma_f32_16x16x32_bf16.
//   per batch b:  out[b][m][n] = sum_k F[m][k] * x[b][n + k - 125]
// Grid: (SEQ/256, 32). Block: 256 threads = 8 waves; wave w owns columns
// [n0 + 32w, n0 + 32w + 32) as TWO 16-wide N-tiles, all 80 filter rows
// (5 M-tiles), K-loop of 8. Each A fragment feeds 2 WMMAs.
// The x window is pre-converted to bf16 in LDS at BOTH parities so every
// B fragment is 8 aligned dword loads (no per-iteration converts).
// ---------------------------------------------------------------------------
__global__ __launch_bounds__(256) void conv_wmma(const float* __restrict__ x,
                                                 const __bf16* __restrict__ filt,
                                                 float* __restrict__ out) {
    __shared__ __align__(16) __bf16 sF[N_FILT * KPAD]; // 40 KB
    __shared__ __align__(16) __bf16 sB0[WIN];          // win[i]   as bf16
    __shared__ __align__(16) __bf16 sB1[WIN];          // win[i+1] as bf16

    const int tid = threadIdx.x;
    const int b   = blockIdx.y;
    const int n0  = blockIdx.x * NTILE_WG;

    // stage filter bank (resident in L2 across the grid) into LDS as b128
    {
        const uint4* fsrc = (const uint4*)filt;
        uint4*       fdst = (uint4*)sF;
        #pragma unroll
        for (int i = tid; i < (N_FILT * KPAD) / 8; i += 256) fdst[i] = fsrc[i];
    }
    // stage x window [n0-125, n0+387) as bf16 at both parities,
    // zero padding outside [0, SEQ)
    {
        const float* xb = x + (size_t)b * SEQ;
        #pragma unroll
        for (int i = tid; i < WIN; i += 256) {
            int n  = n0 - PAD_L + i;
            int n1 = n + 1;
            float v0 = (n  >= 0 && n  < SEQ) ? xb[n]  : 0.0f;
            float v1 = (n1 >= 0 && n1 < SEQ) ? xb[n1] : 0.0f;
            sB0[i] = (__bf16)v0;
            sB1[i] = (__bf16)v1;
        }
    }
    __syncthreads();

    const int lane = tid & 31;
    const int wv   = tid >> 5;     // wave id -> 32-column slab within the chunk
    const int lm   = lane & 15;    // M row (A) / N column (B, C/D)
    const int kg   = lane >> 4;    // lane-half selects K sub-range

    // B fragment source: lane start element p = wv*32 + lm + ks*32 + kg*16.
    // Parity of p == parity of lm; odd lanes read the +1-shifted copy at p-1,
    // so reads are always even-element (dword) aligned.
    const __bf16* sBsel = (lm & 1) ? sB1 : sB0;
    const int     pbase = wv * 32 + (lm & ~1) + kg * 16;

    v8f acc[2][5] = {};            // [n-tile][m-tile] accumulators (80 VGPRs)

    #pragma unroll
    for (int ks = 0; ks < 8; ++ks) {
        // ---- B fragments: 8 aligned LDS dwords each (16 consecutive bf16)
        union { v16bf v; unsigned int u[8]; } t0, t1;
        {
            const unsigned int* q = (const unsigned int*)(sBsel + pbase + ks * 32);
            #pragma unroll
            for (int e = 0; e < 8; ++e) t0.u[e] = q[e];
            #pragma unroll
            for (int e = 0; e < 8; ++e) t1.u[e] = q[8 + e];
        }
        // ---- 5 M-tiles: each A fragment feeds both N-tiles
        #pragma unroll
        for (int mt = 0; mt < 5; ++mt) {
            // A fragment (contiguous K order, same as B): lane half kg, row lm:
            //   elems 0..15 -> K = ks*32 + kg*16 + e
            v16bf afr;
            const __bf16* row = &sF[(size_t)(mt * 16 + lm) * KPAD + ks * 32 + kg * 16];
            #pragma unroll
            for (int e = 0; e < 8; ++e) afr[e] = row[e];
            #pragma unroll
            for (int e = 0; e < 8; ++e) afr[8 + e] = row[8 + e];

            acc[0][mt] = __builtin_amdgcn_wmma_f32_16x16x32_bf16(
                false, afr, false, t0.v, (short)0, acc[0][mt], false, false);
            acc[1][mt] = __builtin_amdgcn_wmma_f32_16x16x32_bf16(
                false, afr, false, t1.v, (short)0, acc[1][mt], false, false);
        }
    }

    // ---- store: C/D layout: VGPR v, lanes 0-15 -> M=v, lanes 16-31 -> M=v+8
    #pragma unroll
    for (int nt = 0; nt < 2; ++nt) {
        const int n = n0 + wv * 32 + nt * 16 + lm;
        float* ob = out + (size_t)b * N_FILT * SEQ + n;
        #pragma unroll
        for (int mt = 0; mt < 5; ++mt) {
            #pragma unroll
            for (int v = 0; v < 8; ++v) {
                int m = mt * 16 + v + 8 * kg;
                ob[(size_t)m * SEQ] = acc[nt][mt][v];
            }
        }
    }
}

extern "C" void kernel_launch(void* const* d_in, const int* in_sizes, int n_in,
                              void* d_out, int out_size, void* d_ws, size_t ws_size,
                              hipStream_t stream) {
    const float* x     = (const float*)d_in[0];   // (32, 1, 32000) f32
    const float* fb1   = (const float*)d_in[1];   // (80,) f32
    const float* fband = (const float*)d_in[2];   // (80,) f32
    float* out = (float*)d_out;                   // (32, 80, 32000) f32

    __bf16* filt = (__bf16*)d_ws;                 // 80*256 bf16 = 40960 B scratch

    gen_filters<<<dim3(N_FILT), dim3(256), 0, stream>>>(fb1, fband, filt);
    conv_wmma<<<dim3(SEQ / NTILE_WG, 32), dim3(256), 0, stream>>>(x, filt, out);
}
